// Model_81097572483333
// MI455X (gfx1250) — matrix operand order repaired and checked by reading, not compiled, not run on hardware
//
#include <hip/hip_runtime.h>
#include <hip/hip_bf16.h>
#include <math.h>

// ---------------------------------------------------------------------------
// Fused Keras-style model on gfx1250 (wave32, WMMA f32_16x16x32_f16):
//   emb-gather -> transpose -> conv1d(w=2,s=2) -> relu -> maxpool(2) ->
//   LSTM(25 steps, U=100, leaky_relu act) -> tanh dense -> softmax [1024,3]
//
// f16 A/B fragments, f32 WMMA accumulate; gate/epilogue math in f32.
// All weights are pre-swizzled by prep_kernel into fragment-major layout
// ([tile][lane][hi], 32 contiguous bytes per lane) so every B fragment is a
// single aligned v16h load (2x global_load_b128) instead of 16 scalar u16s.
// The LSTM's t-invariant weight fragments are register-resident across all
// 25 timesteps (CDNA5: up to 1024 VGPRs/wave).
// ---------------------------------------------------------------------------

typedef __attribute__((ext_vector_type(16))) _Float16 v16h;
typedef __attribute__((ext_vector_type(8)))  _Float16 v8h;
typedef __attribute__((ext_vector_type(8)))  float    v8f;

#define BB     1024   // batch
#define LL     529    // channels after permute (token positions)
#define EE     100    // conv steps (embedding dim)
#define FF     16     // conv filters
#define UU     100    // LSTM units
#define NG     400    // 4*UU gate width
#define KC_PAD 544    // 529 padded up to 17*32 for K-chunking
#define LALPHA 0.2f

// 16-bit A-matrix 16x32 VGPR layout (ISA 7.12.2):
//   lanes 0-15  : row M=lane,    halves 0-7 -> K=0..7,   halves 8-15 -> K=16..23
//   lanes 16-31 : row M=lane-16, halves 0-7 -> K=8..15,  halves 8-15 -> K=24..31
__device__ __forceinline__ int a16_kofs(int lane, int hi) {
  return ((hi < 8) ? 0 : 16) + ((lane < 16) ? 0 : 8) + (hi & 7);
}
// 16-bit B-matrix 32x16 layout: N = lane&15; K order matches A:
//   halves 0-7 -> K=0..7 / 8..15, halves 8-15 -> K=16..23 / 24..31.
__device__ __forceinline__ int b16_kofs(int lane, int hi) {
  return ((hi < 8) ? 0 : 16) + ((lane < 16) ? 0 : 8) + (hi & 7);
}

// ---------------------------------------------------------------------------
// Kernel 0: build fragment-major swizzled f16 weights in workspace.
//   convWswz : [w:2][kc:17][lane:32][hi:16]  <- conv_w[w][c][f], c padded
//   klswz    : [nt:25][lane:32][hi:16]       <- k_lstm[k][n],  K padded 16->32
//   rkswz    : [kc:4][nt:25][lane:32][hi:16] <- rk_lstm[k][n], K padded 100->128
// Each lane's 16 halves are contiguous (32B) and 32B-aligned.
// ---------------------------------------------------------------------------
__global__ void __launch_bounds__(256)
prep_kernel(const float* __restrict__ conv_w, const float* __restrict__ k_lstm,
            const float* __restrict__ rk_lstm,
            _Float16* __restrict__ convWswz, _Float16* __restrict__ klswz,
            _Float16* __restrict__ rkswz) {
  const int nConv = 2 * 17 * 512;      // 17408
  const int nKl   = 25 * 512;          // 12800
  const int nRk   = 4 * 25 * 512;      // 51200
  int i = blockIdx.x * blockDim.x + threadIdx.x;
  if (i < nConv) {
    const int hi = i & 15, lane = (i >> 4) & 31;
    const int kc = (i >> 9) % 17, w = i / (17 * 512);
    const int c = kc * 32 + b16_kofs(lane, hi);
    const int f = lane & 15;
    convWswz[i] = (c < LL) ? (_Float16)conv_w[(w * LL + c) * FF + f] : (_Float16)0.0f;
  } else if (i < nConv + nKl) {
    const int j = i - nConv;
    const int hi = j & 15, lane = (j >> 4) & 31, nt = j >> 9;
    const int k = b16_kofs(lane, hi);
    const int n = nt * 16 + (lane & 15);
    klswz[j] = (k < FF) ? (_Float16)k_lstm[k * NG + n] : (_Float16)0.0f;
  } else if (i < nConv + nKl + nRk) {
    const int j = i - nConv - nKl;
    const int hi = j & 15, lane = (j >> 4) & 31;
    const int nt = (j >> 9) % 25, kc = j / (25 * 512);
    const int k = kc * 32 + b16_kofs(lane, hi);
    const int n = nt * 16 + (lane & 15);
    rkswz[j] = (k < UU) ? (_Float16)rk_lstm[k * NG + n] : (_Float16)0.0f;
  }
}

// ---------------------------------------------------------------------------
// Kernel 1: fused gather + conv1d + bias/ReLU + maxpool.
// One 128-thread workgroup (4 waves) per batch element; wave wv owns conv
// output rows t = wv*16..wv*16+15. K (=channel) chunked by 32 through LDS.
// ---------------------------------------------------------------------------
__global__ void __launch_bounds__(128)
conv_pool_kernel(const int* __restrict__ tokens, const float* __restrict__ emb,
                 const _Float16* __restrict__ convWswz, const float* __restrict__ convB,
                 _Float16* __restrict__ pooled) {
  __shared__ _Float16 chL[100 * 40];   // [step s][chunk-channel c'] (stride 40)

  const int b    = blockIdx.x;
  const int tid  = threadIdx.x;
  const int lane = tid & 31;
  const int wv   = tid >> 5;           // 0..3 -> M tile t0 = wv*16
  const int t0   = wv * 16;
  const int arow = lane & 15;          // output step within tile (A row)
  const int nf   = lane & 15;          // filter column
  v8f accE = {};                       // even-tap accumulator (x[2t]   * W0)
  v8f accO = {};                       // odd-tap  accumulator (x[2t+1] * W1)

  // A-fragment LDS addresses are chunk-invariant (2x16B contiguous runs)
  int sE = 2 * (t0 + arow);
  int sO = sE + 1;
  if (sE > 99) sE = 99;                // rows t>=50 are garbage; masked at store
  if (sO > 99) sO = 99;
  const int aBaseLo = ((lane < 16) ? 0 : 8);       // halves 0-7 -> K 0..7 / 8..15
  const int aBaseHi = 16 + aBaseLo;                // halves 8-15 -> K 16..23 / 24..31

  for (int kc = 0; kc < 17; ++kc) {
    // --- cooperative gather: 32 channels x 100 steps, f32 -> f16, transposed.
    // Index-clamped so loads are unconditional (no exec-mask branch ladder);
    // only the stored value is selected.
    {
      const int cl    = tid & 31;
      const int c     = kc * 32 + cl;
      const bool valid = (c < LL);
      const int tok   = tokens[b * LL + (valid ? c : (LL - 1))];
      const int sseg  = tid >> 5;      // 4 segments of 25 steps
      for (int s = sseg * 25; s < sseg * 25 + 25; ++s) {
        const float v = emb[tok * EE + s];
        chL[s * 40 + cl] = valid ? (_Float16)v : (_Float16)0.0f;
      }
    }
    __syncthreads();

    // --- A fragments (even/odd taps): vectorized ds_load_b128 pairs
    v16h aE, aO;
    {
      const v8h eLo = *(const v8h*)&chL[sE * 40 + aBaseLo];
      const v8h eHi = *(const v8h*)&chL[sE * 40 + aBaseHi];
      const v8h oLo = *(const v8h*)&chL[sO * 40 + aBaseLo];
      const v8h oHi = *(const v8h*)&chL[sO * 40 + aBaseHi];
      for (int i = 0; i < 8; ++i) {
        aE[i] = eLo[i]; aE[i + 8] = eHi[i];
        aO[i] = oLo[i]; aO[i + 8] = oHi[i];
      }
    }
    // --- B fragments: single aligned v16h load each (fragment-major layout)
    const v16h bw0 = *(const v16h*)&convWswz[((0 * 17 + kc) * 32 + lane) * 16];
    const v16h bw1 = *(const v16h*)&convWswz[((1 * 17 + kc) * 32 + lane) * 16];

    accE = __builtin_amdgcn_wmma_f32_16x16x32_f16(false, aE, false, bw0, (short)0, accE, false, false);
    accO = __builtin_amdgcn_wmma_f32_16x16x32_f16(false, aO, false, bw1, (short)0, accO, false, false);
    __syncthreads();                   // chL is rewritten next chunk
  }

  // --- bias + ReLU + 2x maxpool; D layout: VGPR r <-> row M = r + 8*(lane>=16)
  const float cb = convB[nf];
  for (int q = 0; q < 4; ++q) {
    const float y0 = fmaxf(accE[2 * q]     + accO[2 * q]     + cb, 0.0f);
    const float y1 = fmaxf(accE[2 * q + 1] + accO[2 * q + 1] + cb, 0.0f);
    const float pv = fmaxf(y0, y1);
    const int p = wv * 8 + ((lane < 16) ? 0 : 4) + q;   // pooled step index
    if (p < 25) pooled[(b * 25 + p) * FF + nf] = (_Float16)pv;
  }
}

// ---------------------------------------------------------------------------
// Kernel 2: LSTM (25 steps) + tanh dense + softmax, fully on-chip.
// One 256-thread workgroup (8 waves) per 16 batch rows. Wave wv owns gate
// N-tiles nt = wv + 8q (q<4, nt<25). All weight B fragments (1 k_lstm +
// 4 rk_lstm K-chunks per N-tile) are loaded ONCE into VGPRs before the time
// loop; the recurrence inner loop is pure LDS(h) -> WMMA -> LDS(z).
// ---------------------------------------------------------------------------
__global__ void __launch_bounds__(256)
lstm_head_kernel(const _Float16* __restrict__ pooled,
                 const _Float16* __restrict__ klswz, const _Float16* __restrict__ rkswz,
                 const float* __restrict__ blstm,
                 const float* __restrict__ w1, const float* __restrict__ b1,
                 const float* __restrict__ w2, const float* __restrict__ b2,
                 float* __restrict__ out) {
  __shared__ _Float16 h16s[16 * 128];  // f16 hidden state, K-padded (cols 100..127 = 0)
  __shared__ float    zS[16 * NG];     // pre-activation gates for this step
  __shared__ float    hS[16 * UU];     // f32 hidden state (for dense head)
  __shared__ float    cS[16 * UU];     // cell state
  __shared__ float    dS[16 * 64];     // tanh dense activations
  __shared__ float    lgS[16 * 3];     // logits

  const int tid  = threadIdx.x;
  const int lane = tid & 31;
  const int wv   = tid >> 5;           // 0..7
  const int b0   = blockIdx.x * 16;
  const int arow = lane & 15;          // batch row within tile (A rows)
  const int ncol = lane & 15;          // gate column within N tile
  const int moff = (lane < 16) ? 0 : 8;

  for (int i = tid; i < 16 * 128; i += 256) h16s[i] = (_Float16)0.0f;
  for (int i = tid; i < 16 * UU;  i += 256) cS[i] = 0.0f;

  // ---- preload t-invariant weight fragments into registers ----------------
  v16h bk[4];          // k_lstm fragment per owned N tile
  v16h br[4][4];       // rk_lstm fragment per owned N tile x K chunk
  float bias[4];
#pragma unroll
  for (int q = 0; q < 4; ++q) {
    const int nt = wv + q * 8;
    if (nt < 25) {                     // wave-uniform guard
      bk[q] = *(const v16h*)&klswz[(nt * 32 + lane) * 16];
#pragma unroll
      for (int kc = 0; kc < 4; ++kc)
        br[q][kc] = *(const v16h*)&rkswz[((kc * 25 + nt) * 32 + lane) * 16];
      bias[q] = blstm[nt * 16 + ncol];
    }
  }

  for (int t = 0; t < 25; ++t) {
    __syncthreads();                   // h16s/zS stable across phases

    // ---- GEMM phase -------------------------------------------------------
    // input A fragment: 8 contiguous valid halves (K 0..7 / 8..15) + 8 zeros
    v16h ain;
    {
      const v8h lo = *(const v8h*)&pooled[((b0 + arow) * 25 + t) * FF + moff];
      for (int i = 0; i < 8; ++i) { ain[i] = lo[i]; ain[i + 8] = (_Float16)0.0f; }
    }
    // hidden-state A fragments (shared across this wave's N tiles)
    v16h ah[4];
#pragma unroll
    for (int kc = 0; kc < 4; ++kc) {
      const v8h hLo = *(const v8h*)&h16s[arow * 128 + kc * 32 + moff];
      const v8h hHi = *(const v8h*)&h16s[arow * 128 + kc * 32 + 16 + moff];
      for (int i = 0; i < 8; ++i) { ah[kc][i] = hLo[i]; ah[kc][i + 8] = hHi[i]; }
    }
#pragma unroll
    for (int q = 0; q < 4; ++q) {
      const int nt = wv + q * 8;
      if (nt < 25) {                   // wave-uniform: EXEC stays all-ones
        v8f acc = {};
        acc = __builtin_amdgcn_wmma_f32_16x16x32_f16(false, ain,   false, bk[q],    (short)0, acc, false, false);
#pragma unroll
        for (int kc = 0; kc < 4; ++kc)
          acc = __builtin_amdgcn_wmma_f32_16x16x32_f16(false, ah[kc], false, br[q][kc], (short)0, acc, false, false);
        const int n = nt * 16 + ncol;
        for (int r = 0; r < 8; ++r)
          zS[(r + moff) * NG + n] = acc[r] + bias[q];
      }
    }
    __syncthreads();

    // ---- gate phase (i,f,g,o; act = leaky_relu, recurrent_act = sigmoid) --
    for (int idx = tid; idx < 16 * UU; idx += 256) {
      const int row = idx / UU, u = idx - row * UU;
      const float zi = zS[row * NG + u];
      const float zf = zS[row * NG + UU + u];
      const float zg = zS[row * NG + 2 * UU + u];
      const float zo = zS[row * NG + 3 * UU + u];
      const float ig = 1.0f / (1.0f + expf(-zi));
      const float fg = 1.0f / (1.0f + expf(-zf));
      const float og = 1.0f / (1.0f + expf(-zo));
      const float g  = (zg >= 0.0f) ? zg : LALPHA * zg;
      const float c  = fg * cS[idx] + ig * g;
      cS[idx] = c;
      const float lc = (c >= 0.0f) ? c : LALPHA * c;
      const float h  = og * lc;
      hS[idx] = h;
      h16s[row * 128 + u] = (_Float16)h;
    }
  }
  __syncthreads();

  // ---- dense head: d = tanh(h @ w1 + b1) ---------------------------------
  for (int idx = tid; idx < 16 * 64; idx += 256) {
    const int row = idx >> 6, j = idx & 63;
    float s = b1[j];
    for (int u = 0; u < UU; ++u) s += hS[row * UU + u] * w1[u * 64 + j];
    dS[idx] = tanhf(s);
  }
  __syncthreads();
  if (tid < 48) {
    const int row = tid / 3, k = tid - row * 3;
    float s = b2[k];
    for (int j = 0; j < 64; ++j) s += dS[row * 64 + j] * w2[j * 3 + k];
    lgS[row * 3 + k] = s;
  }
  __syncthreads();
  if (tid < 16) {
    const float v0 = lgS[tid * 3], v1 = lgS[tid * 3 + 1], v2 = lgS[tid * 3 + 2];
    const float m  = fmaxf(v0, fmaxf(v1, v2));
    const float e0 = expf(v0 - m), e1 = expf(v1 - m), e2 = expf(v2 - m);
    const float inv = 1.0f / (e0 + e1 + e2);
    out[(b0 + tid) * 3 + 0] = e0 * inv;
    out[(b0 + tid) * 3 + 1] = e1 * inv;
    out[(b0 + tid) * 3 + 2] = e2 * inv;
  }
}

// ---------------------------------------------------------------------------
// Workspace layout (bytes, 256-aligned):
//   [0,       34816)  convWswz  f16[2][17][32][16]
//   [34816,   60416)  klswz     f16[25][32][16]
//   [60416,  162816)  rkswz     f16[4][25][32][16]
//   [162816, 982016)  pooled    f16[1024][25][16]
// ---------------------------------------------------------------------------
extern "C" void kernel_launch(void* const* d_in, const int* in_sizes, int n_in,
                              void* d_out, int out_size, void* d_ws, size_t ws_size,
                              hipStream_t stream) {
  const int*   tokens  = (const int*)  d_in[0];
  const float* emb     = (const float*)d_in[1];
  const float* conv_w  = (const float*)d_in[2];
  const float* conv_b  = (const float*)d_in[3];
  const float* k_lstm  = (const float*)d_in[4];
  const float* rk_lstm = (const float*)d_in[5];
  const float* b_lstm  = (const float*)d_in[6];
  const float* w1      = (const float*)d_in[7];
  const float* b1      = (const float*)d_in[8];
  const float* w2      = (const float*)d_in[9];
  const float* b2      = (const float*)d_in[10];
  float* out = (float*)d_out;

  char* ws = (char*)d_ws;
  _Float16* convWswz = (_Float16*)(ws + 0);
  _Float16* klswz    = (_Float16*)(ws + 34816);
  _Float16* rkswz    = (_Float16*)(ws + 60416);
  _Float16* pooled   = (_Float16*)(ws + 162816);

  const int prepTot = 2 * 17 * 512 + 25 * 512 + 4 * 25 * 512;
  prep_kernel<<<(prepTot + 255) / 256, 256, 0, stream>>>(conv_w, k_lstm, rk_lstm,
                                                         convWswz, klswz, rkswz);
  conv_pool_kernel<<<BB, 128, 0, stream>>>(tokens, emb, convWswz, conv_b, pooled);
  lstm_head_kernel<<<BB / 16, 256, 0, stream>>>(pooled, klswz, rkswz, b_lstm,
                                                w1, b1, w2, b2, out);
}
